// RelatEntAtt_18734647345345
// MI455X (gfx1250) — matrix operand order repaired and checked by reading, not compiled, not run on hardware
//
#include <hip/hip_runtime.h>
#include <hip/hip_bf16.h>
#include <math.h>

// Problem constants (fixed by the reference).
constexpr int E = 4096;
constexpr int R = 64;
constexpr int D = 256;
constexpr int CH = E / 1024;        // e-chunks per 1024-thread block
constexpr float LRELU = 0.2f;

typedef __attribute__((ext_vector_type(16))) _Float16 v16h;
typedef __attribute__((ext_vector_type(8)))  float    v8f;

struct alignas(16) F4 { float v[4]; };   // 16B vector load unit -> global_load_b128

// ---------------- helpers ----------------
__device__ __forceinline__ float wred_max(float v) {
#pragma unroll
  for (int o = 16; o > 0; o >>= 1) v = fmaxf(v, __shfl_xor(v, o, 32));
  return v;
}
__device__ __forceinline__ float wred_sum(float v) {
#pragma unroll
  for (int o = 16; o > 0; o >>= 1) v += __shfl_xor(v, o, 32);
  return v;
}
// float atomic-max on LDS via signed/unsigned integer ordering trick.
__device__ __forceinline__ void atomicMaxF_shared(float* a, float v) {
  if (v >= 0.f) atomicMax(reinterpret_cast<int*>(a), __float_as_int(v));
  else          atomicMin(reinterpret_cast<unsigned int*>(a), __float_as_uint(v));
}
__device__ __forceinline__ float eluf(float x) { return x > 0.f ? x : (__expf(x) - 1.f); }

// ---------------- Kernel A: adj = softmax(adj_agg, axis=1) ----------------
__global__ void adj_softmax_kernel(const float* __restrict__ adj_agg,
                                   float* __restrict__ adj) {
  __shared__ float red[R];
  const int e = blockIdx.x, t = threadIdx.x;
  const float v = adj_agg[e * R + t];
  red[t] = v; __syncthreads();
  for (int s = R / 2; s > 0; s >>= 1) { if (t < s) red[t] = fmaxf(red[t], red[t + s]); __syncthreads(); }
  const float m = red[0]; __syncthreads();
  const float ev = __expf(v - m);
  red[t] = ev; __syncthreads();
  for (int s = R / 2; s > 0; s >>= 1) { if (t < s) red[t] += red[t + s]; __syncthreads(); }
  adj[e * R + t] = ev / red[0];
}

// ---------------- Kernel B: per-d column — all softmax stats + h', r' ----------------
// One block per d. Three streaming passes over the virtual att[:, :, d] slab:
//   pass1: M_R[e,d] (registers->ws), M_E[j,d] (LDS atomic max)
//   pass2: S_R[e,d] (->ws as reciprocal), S_E[j,d] (LDS atomic add)
//   pass3: T_E[e,d]=sum_j att_E -> h', RSum[j,d]=sum_e att_R -> r'
__global__ void att_core_kernel(const float* __restrict__ h, const float* __restrict__ r,
                                const float* __restrict__ adj,
                                float* __restrict__ M_Rg, float* __restrict__ iSRg,
                                float* __restrict__ out_h, float* __restrict__ out_r) {
  const int d = blockIdx.x;
  const int t = threadIdx.x;           // 0..1023
  const int lane = t & 31;
  __shared__ float rS[R], ME[R], SE[R], RSum[R];
  if (t < R) { rS[t] = r[t * D + d]; ME[t] = -INFINITY; SE[t] = 0.f; RSum[t] = 0.f; }
  __syncthreads();

  float hd[CH], mr[CH], isr[CH];
#pragma unroll
  for (int c = 0; c < CH; ++c) {
    const int e = t + 1024 * c;
    hd[c] = h[e * D + d];
    const float* arow = adj + e * R;
    __builtin_prefetch(arow, 0, 0);                 // global_prefetch_b8
    float mloc = -INFINITY;
    for (int j = 0; j < R; ++j) {
      const float tt = arow[j] * hd[c] * rS[j];
      const float x = tt >= 0.f ? tt : LRELU * tt;  // LeakyReLU
      mloc = fmaxf(mloc, x);
      const float wm = wred_max(x);                 // j is wave-uniform
      if (lane == 0) atomicMaxF_shared(&ME[j], wm);
    }
    mr[c] = mloc;
    M_Rg[e * D + d] = mloc;
  }
  __syncthreads();

#pragma unroll
  for (int c = 0; c < CH; ++c) {
    const int e = t + 1024 * c;
    const float* arow = adj + e * R;
    float s = 0.f;
    for (int j = 0; j < R; ++j) {
      const float tt = arow[j] * hd[c] * rS[j];
      const float x = tt >= 0.f ? tt : LRELU * tt;
      s += __expf(x - mr[c]);
      const float se = wred_sum(__expf(x - ME[j]));
      if (lane == 0) atomicAdd(&SE[j], se);
    }
    isr[c] = 1.f / s;
    iSRg[e * D + d] = isr[c];
  }
  __syncthreads();
  if (t < R) SE[t] = 1.f / SE[t];      // invert once
  __syncthreads();

#pragma unroll
  for (int c = 0; c < CH; ++c) {
    const int e = t + 1024 * c;
    const float* arow = adj + e * R;
    float te = 0.f;
    for (int j = 0; j < R; ++j) {
      const float tt = arow[j] * hd[c] * rS[j];
      const float x = tt >= 0.f ? tt : LRELU * tt;
      const float aR = __expf(x - mr[c]) * isr[c];  // att_R
      const float wr = wred_sum(aR);
      if (lane == 0) atomicAdd(&RSum[j], wr);
      te += __expf(x - ME[j]) * SE[j];              // att_E contribution
    }
    out_h[e * D + d] = eluf(hd[c] * te);            // elu(h')
  }
  __syncthreads();
  if (t < R) out_r[t * D + d] = eluf(rS[t] * RSum[t]);   // elu(r')
}

// ---------------- Kernel C: lin[e,j] = att_R[e,j,:]·W + b via v_wmma_f32_16x16x32_f16 ----
// tile = 16 consecutive (e,j) rows (fixed e, 16 consecutive j). A is built in-register
// per the 16-bit A 16x32 layout; B has W in column N=0 only; f32 accumulation.
// Per lane group g, the 16 A-fragment K values are two contiguous 8-float runs
// [d0+8g, d0+8g+8) and [d0+16+8g, d0+24+8g)  =>  4x b128 loads per input array.
__global__ void lin_wmma_kernel(const float* __restrict__ adj, const float* __restrict__ h,
                                const float* __restrict__ r,
                                const float* __restrict__ M_Rg, const float* __restrict__ iSRg,
                                const float* __restrict__ W, const float* __restrict__ bl,
                                float* __restrict__ out_a) {
  const int lane = threadIdx.x & 31;
  const int tile = blockIdx.x * (blockDim.x >> 5) + (threadIdx.x >> 5);
  const int e  = tile >> 2;            // 4 tiles of 16 j per e (R=64)
  const int j0 = (tile & 3) << 4;
  const int g  = lane >> 4;            // lane half-group
  const int m  = lane & 15;            // A-matrix row within tile
  const int j  = j0 + m;
  const float adjv = adj[e * R + j];
  const float wmask = (m == 0) ? 1.f : 0.f;   // select B column 0 by value, no branch

  const F4* h4 = reinterpret_cast<const F4*>(h    + e * D);
  const F4* r4 = reinterpret_cast<const F4*>(r    + j * D);
  const F4* m4 = reinterpret_cast<const F4*>(M_Rg + e * D);
  const F4* s4 = reinterpret_cast<const F4*>(iSRg + e * D);
  const F4* w4 = reinterpret_cast<const F4*>(W);

  v8f acc = {};
  for (int d0 = 0; d0 < D; d0 += 32) {
    // vectorized fragment fetches: run0 at d0+8g, run1 at d0+16+8g (in F4 units)
    const int b0 = (d0 + 8 * g) >> 2;
    const int b1 = (d0 + 16 + 8 * g) >> 2;
    F4 hv[4], rv[4], mv[4], sv[4], wv[4];
#pragma unroll
    for (int q = 0; q < 2; ++q) {
      hv[q] = h4[b0 + q]; hv[2 + q] = h4[b1 + q];
      rv[q] = r4[b0 + q]; rv[2 + q] = r4[b1 + q];
      mv[q] = m4[b0 + q]; mv[2 + q] = m4[b1 + q];
      sv[q] = s4[b0 + q]; sv[2 + q] = s4[b1 + q];
    }
#pragma unroll
    for (int q = 0; q < 2; ++q) { wv[q] = w4[b0 + q]; wv[2 + q] = w4[b1 + q]; }

    v16h a, bmat;
#pragma unroll
    for (int i = 0; i < 16; ++i) {
      const int q = i >> 2, k = i & 3;          // fetched element (i>>3 picks run)
      const float tt = adjv * hv[q].v[k] * rv[q].v[k];
      const float x  = tt >= 0.f ? tt : LRELU * tt;
      const float aR = __expf(x - mv[q].v[k]) * sv[q].v[k];   // att_R in-register
      a[i] = (_Float16)aR;
      bmat[i] = (_Float16)(wmask * wv[q].v[k]); // v_cndmask-free: multiply-select
    }
    acc = __builtin_amdgcn_wmma_f32_16x16x32_f16(false, a, false, bmat,
                                                 (short)0, acc, false, false);
  }
  if (m == 0) {                        // column N=0 lives in lanes 0 and 16
    const float bb = bl[0];
#pragma unroll
    for (int v = 0; v < 8; ++v)
      out_a[tile * 16 + 8 * g + v] = acc[v] + bb;   // D layout: M = v + 8*g
  }
}

// ---------------- Kernel D: in-place grouped softmax of alpha ----------------
// reshape(R,E) + softmax(axis=1) == softmax over contiguous E-element chunks of lin.
__global__ void alpha_softmax_kernel(float* __restrict__ out_a) {
  __shared__ float sred[32];
  const int t = threadIdx.x, lane = t & 31, wid = t >> 5;
  float* base = out_a + (size_t)blockIdx.x * E;
  float v[CH];
  float m = -INFINITY;
#pragma unroll
  for (int c = 0; c < CH; ++c) { v[c] = base[t + 1024 * c]; m = fmaxf(m, v[c]); }
  m = wred_max(m);
  if (lane == 0) sred[wid] = m;
  __syncthreads();
  if (t < 32) { float x = wred_max(sred[t]); if (t == 0) sred[0] = x; }
  __syncthreads();
  m = sred[0];
  __syncthreads();
  float s = 0.f, ev[CH];
#pragma unroll
  for (int c = 0; c < CH; ++c) { ev[c] = __expf(v[c] - m); s += ev[c]; }
  s = wred_sum(s);
  if (lane == 0) sred[wid] = s;
  __syncthreads();
  if (t < 32) { float x = wred_sum(sred[t]); if (t == 0) sred[0] = x; }
  __syncthreads();
  const float inv = 1.f / sred[0];
#pragma unroll
  for (int c = 0; c < CH; ++c) base[t + 1024 * c] = ev[c] * inv;
}

// ---------------- launcher ----------------
extern "C" void kernel_launch(void* const* d_in, const int* in_sizes, int n_in,
                              void* d_out, int out_size, void* d_ws, size_t ws_size,
                              hipStream_t stream) {
  const float* h    = (const float*)d_in[0];   // (E,D)
  const float* r    = (const float*)d_in[1];   // (R,D)
  const float* adjg = (const float*)d_in[2];   // (E,R)
  const float* W    = (const float*)d_in[3];   // (D,1)
  const float* b    = (const float*)d_in[4];   // (1,)

  float* ws   = (float*)d_ws;                  // 9.0 MiB used
  float* adj  = ws;                            // E*R
  float* M_Rg = adj  + (size_t)E * R;          // E*D
  float* iSRg = M_Rg + (size_t)E * D;          // E*D

  float* out   = (float*)d_out;                // concat: elu(h'), elu(r'), alpha
  float* out_h = out;
  float* out_r = out_h + (size_t)E * D;
  float* out_a = out_r + (size_t)R * D;

  adj_softmax_kernel  <<<E, R,    0, stream>>>(adjg, adj);
  att_core_kernel     <<<D, 1024, 0, stream>>>(h, r, adj, M_Rg, iSRg, out_h, out_r);
  lin_wmma_kernel     <<<(E * R / 16) / 8, 256, 0, stream>>>(adj, h, r, M_Rg, iSRg, W, b, out_a);
  alpha_softmax_kernel<<<R, 1024, 0, stream>>>(out_a);
}
